// MultiheadAttention_22531398435152
// MI455X (gfx1250) — matrix operand order repaired and checked by reading, not compiled, not run on hardware
//
#include <hip/hip_runtime.h>
#include <hip/hip_bf16.h>
#include <math.h>

// ---------------- problem constants ----------------
#define DMODEL 1024
#define NHEAD  16
#define HDIM   64
#define SEQL   2048
#define BATCH  2
#define MTOK   (BATCH * SEQL)          // 4096 tokens total
#define BH     (BATCH * NHEAD)         // 32

typedef __bf16 bf16_t;
typedef __attribute__((ext_vector_type(16))) __bf16 v16bf;
typedef __attribute__((ext_vector_type(8)))  __bf16 v8bf;
typedef __attribute__((ext_vector_type(8)))  float  v8f;
typedef __attribute__((ext_vector_type(4)))  float  v4f;

union ABFrag { v16bf v; struct { v8bf lo, hi; } h; };

// ----- A fragment: 16x32 bf16, lane = row M (&15), half = lane>>4 -----
// elements 0..7  -> K = k0 + half*8 + e
// elements 8..15 -> K = k0 + 16 + half*8 + (e-8)
__device__ __forceinline__ v16bf load_a_frag(const bf16_t* arow, int k0, int half) {
  ABFrag u;
  u.h.lo = *(const v8bf*)(arow + k0 + half * 8);
  u.h.hi = *(const v8bf*)(arow + k0 + 16 + half * 8);
  return u.v;
}

// ----- B fragment from B^T (row-major [N,K]): lane = col N, same K order as A fragment -----
__device__ __forceinline__ v16bf load_b_frag(const bf16_t* brow, int k0, int half) {
  ABFrag u;
  u.h.lo = *(const v8bf*)(brow + k0 + half * 8);
  u.h.hi = *(const v8bf*)(brow + k0 + 16 + half * 8);
  return u.v;
}

// ----- A fragment built from fp32 source (attn rows), cvt to bf16 in regs -----
__device__ __forceinline__ v16bf load_a_frag_f32(const float* arow, int k0, int half) {
  const float* p0 = arow + k0 + half * 8;
  const float* p1 = arow + k0 + 16 + half * 8;
  v4f a0 = *(const v4f*)(p0);
  v4f a1 = *(const v4f*)(p0 + 4);
  v4f b0 = *(const v4f*)(p1);
  v4f b1 = *(const v4f*)(p1 + 4);
  v16bf a;
#pragma unroll
  for (int i = 0; i < 4; ++i) {
    a[i]      = (bf16_t)a0[i];
    a[i + 4]  = (bf16_t)a1[i];
    a[i + 8]  = (bf16_t)b0[i];
    a[i + 12] = (bf16_t)b1[i];
  }
  return a;
}

__device__ __forceinline__ v8f wmma_acc(v16bf a, v16bf b, v8f c) {
  return __builtin_amdgcn_wmma_f32_16x16x32_bf16(false, a, false, b, (short)0, c,
                                                 false, false);
}

// 2x2 register-blocked: wave computes a 32x32 fp32 tile.
// A row-major [.,lda] bf16 at rows m0 / m0+16 ; BT row-major [.,ldb] bf16 at
// rows n0 / n0+16. acc[0]=(0,0) acc[1]=(0,1) acc[2]=(1,0) acc[3]=(1,1).
__device__ __forceinline__ void wmma_tile_2x2(const bf16_t* A, int lda,
                                              const bf16_t* BT, int ldb,
                                              int kdim, int lane, v8f acc[4]) {
  const int r = lane & 15, half = lane >> 4;
  const bf16_t* a0 = A + (size_t)r * lda;
  const bf16_t* a1 = a0 + (size_t)16 * lda;
  const bf16_t* b0 = BT + (size_t)r * ldb;
  const bf16_t* b1 = b0 + (size_t)16 * ldb;
  for (int k0 = 0; k0 < kdim; k0 += 32) {
    if (k0 + 64 < kdim) {
      __builtin_prefetch(a0 + k0 + 64, 0, 3);
      __builtin_prefetch(a1 + k0 + 64, 0, 3);
      __builtin_prefetch(b0 + k0 + 64, 0, 3);
      __builtin_prefetch(b1 + k0 + 64, 0, 3);
    }
    v16bf fa0 = load_a_frag(a0, k0, half);
    v16bf fa1 = load_a_frag(a1, k0, half);
    v16bf fb0 = load_b_frag(b0, k0, half);
    v16bf fb1 = load_b_frag(b1, k0, half);
    acc[0] = wmma_acc(fa0, fb0, acc[0]);
    acc[1] = wmma_acc(fa0, fb1, acc[1]);
    acc[2] = wmma_acc(fa1, fb0, acc[2]);
    acc[3] = wmma_acc(fa1, fb1, acc[3]);
  }
}

// ---------------- stage 0: fp32 -> bf16 ----------------
__global__ void cvt_bf16_kernel(const float* __restrict__ src,
                                bf16_t* __restrict__ dst, int n) {
  int i = (blockIdx.x * blockDim.x + threadIdx.x) * 4;
  if (i + 3 < n) {
    v4f v = *(const v4f*)(src + i);
    dst[i + 0] = (bf16_t)v[0];
    dst[i + 1] = (bf16_t)v[1];
    dst[i + 2] = (bf16_t)v[2];
    dst[i + 3] = (bf16_t)v[3];
  }
}

// ---------------- stage 0b: W [K,N] fp32 -> W^T [N,K] bf16 (LDS transpose) ----
__global__ void cvt_transpose_kernel(const float* __restrict__ W,
                                     bf16_t* __restrict__ WT) {
  __shared__ float tile[32][33];
  const int bx = blockIdx.x * 32;  // n
  const int by = blockIdx.y * 32;  // k
  const int tx = threadIdx.x, ty = threadIdx.y;  // 32 x 8
#pragma unroll
  for (int j = 0; j < 32; j += 8)
    tile[ty + j][tx] = W[(size_t)(by + ty + j) * DMODEL + (bx + tx)];
  __syncthreads();
#pragma unroll
  for (int j = 0; j < 32; j += 8)
    WT[(size_t)(bx + ty + j) * DMODEL + (by + tx)] = (bf16_t)tile[tx][ty + j];
}

// ---------------- stage 1: fused QKV projection -------------------------
// grid (DMODEL/128, MTOK/32, 3), block 128 (4 waves, each a 32x32 tile)
__global__ void qkv_proj_kernel(const bf16_t* __restrict__ xb,
                                const bf16_t* __restrict__ wqT,
                                const bf16_t* __restrict__ wkT,
                                const bf16_t* __restrict__ wvT,
                                const float* __restrict__ bq,
                                const float* __restrict__ bk,
                                const float* __restrict__ bv,
                                bf16_t* __restrict__ Qb,
                                bf16_t* __restrict__ Kb,
                                bf16_t* __restrict__ Vt) {
  const int lane = threadIdx.x & 31;
  const int wave = threadIdx.x >> 5;
  const int n0 = (blockIdx.x * 4 + wave) * 32;  // 0..992
  const int m0 = blockIdx.y * 32;               // 0..4064
  const int which = blockIdx.z;                 // 0:Q 1:K 2:V
  const bf16_t* WT = (which == 0) ? wqT : (which == 1) ? wkT : wvT;
  const float* bias = (which == 0) ? bq : (which == 1) ? bk : bv;
  bf16_t* rowdst = (which == 0) ? Qb : Kb;

  v8f acc[4] = {v8f{}, v8f{}, v8f{}, v8f{}};
  wmma_tile_2x2(xb + (size_t)m0 * DMODEL, DMODEL,
                WT + (size_t)n0 * DMODEL, DMODEL, DMODEL, lane, acc);

  const int half = lane >> 4, col = lane & 15;
#pragma unroll
  for (int i = 0; i < 2; ++i) {
#pragma unroll
    for (int j = 0; j < 2; ++j) {
      const v8f a = acc[i * 2 + j];
      const int n = n0 + j * 16 + col;  // 0..1023
      const int h = n >> 6, d = n & 63;
      const float bz = bias[n];
#pragma unroll
      for (int r = 0; r < 8; ++r) {
        const int m = m0 + i * 16 + r + half * 8;  // global token
        const int b = m >> 11, l = m & 2047;
        const float val = a[r] + bz;
        if (which == 2) {
          // V stored transposed: Vt[b,h,d,l]
          Vt[(((size_t)(b * NHEAD + h)) * HDIM + d) * SEQL + l] = (bf16_t)val;
        } else {
          // Q/K stored [b,h,l,d]
          rowdst[(((size_t)(b * NHEAD + h)) * SEQL + l) * HDIM + d] = (bf16_t)val;
        }
      }
    }
  }
}

// ---------------- stage 2: scores = Q K^T / sqrt(d) --------------------
// grid (SEQL/128, SEQL/32, BH), block 128 (4 waves, each a 32x32 tile)
__global__ void scores_kernel(const bf16_t* __restrict__ Qb,
                              const bf16_t* __restrict__ Kb,
                              float* __restrict__ attn) {
  const int lane = threadIdx.x & 31;
  const int wave = threadIdx.x >> 5;
  const int n0 = (blockIdx.x * 4 + wave) * 32;  // 0..2016
  const int m0 = blockIdx.y * 32;               // 0..2016
  const int bh = blockIdx.z;                    // 0..31

  const bf16_t* A  = Qb + ((size_t)bh * SEQL + m0) * HDIM;
  const bf16_t* BT = Kb + ((size_t)bh * SEQL + n0) * HDIM;
  v8f acc[4] = {v8f{}, v8f{}, v8f{}, v8f{}};
  wmma_tile_2x2(A, HDIM, BT, HDIM, HDIM, lane, acc);

  const int half = lane >> 4, col = lane & 15;
  float* out = attn + ((size_t)bh * SEQL + m0) * SEQL + n0;
#pragma unroll
  for (int i = 0; i < 2; ++i) {
#pragma unroll
    for (int j = 0; j < 2; ++j) {
      const v8f a = acc[i * 2 + j];
#pragma unroll
      for (int r = 0; r < 8; ++r)
        out[(size_t)(i * 16 + r + half * 8) * SEQL + j * 16 + col] =
            a[r] * 0.125f;  // 1/sqrt(64)
    }
  }
}

// ---------------- stage 3: row softmax in place -------------------------
// one 256-thread block per row of 2048
__global__ void softmax_kernel(float* __restrict__ attn) {
  float* p = attn + (size_t)blockIdx.x * SEQL;
  const int tid = threadIdx.x;
  const int wv = tid >> 5, ln = tid & 31;
  __shared__ float redmax[8];
  __shared__ float redsum[8];

  float v[8];
  float m = -3.0e38f;
#pragma unroll
  for (int i = 0; i < 8; ++i) {
    v[i] = p[tid + i * 256];
    m = fmaxf(m, v[i]);
  }
#pragma unroll
  for (int off = 16; off > 0; off >>= 1) m = fmaxf(m, __shfl_xor(m, off, 32));
  if (ln == 0) redmax[wv] = m;
  __syncthreads();
  float mall = redmax[0];
#pragma unroll
  for (int i = 1; i < 8; ++i) mall = fmaxf(mall, redmax[i]);

  float s = 0.f;
#pragma unroll
  for (int i = 0; i < 8; ++i) {
    v[i] = __expf(v[i] - mall);
    s += v[i];
  }
#pragma unroll
  for (int off = 16; off > 0; off >>= 1) s += __shfl_xor(s, off, 32);
  if (ln == 0) redsum[wv] = s;
  __syncthreads();
  float sall = 0.f;
#pragma unroll
  for (int i = 0; i < 8; ++i) sall += redsum[i];
  const float inv = 1.0f / sall;
#pragma unroll
  for (int i = 0; i < 8; ++i) p[tid + i * 256] = v[i] * inv;
}

// ---------------- stage 4: ctx = P V  (A fp32 -> bf16 in regs) ----------
// grid (SEQL/128, BH), block 128; each wave: 2 M-tiles x all 4 N-tiles (2x4)
__global__ void ctx_gemm_kernel(const float* __restrict__ attn,
                                const bf16_t* __restrict__ Vt,
                                bf16_t* __restrict__ ctx) {
  const int lane = threadIdx.x & 31;
  const int wave = threadIdx.x >> 5;
  const int m0 = (blockIdx.x * 4 + wave) * 32;  // 0..2016
  const int bh = blockIdx.y;                    // 0..31
  const int r = lane & 15, half = lane >> 4;

  const float* arow0 = attn + ((size_t)bh * SEQL + m0 + r) * SEQL;
  const float* arow1 = arow0 + (size_t)16 * SEQL;
  const bf16_t* b0 = Vt + ((size_t)bh * HDIM + r) * SEQL;
  const bf16_t* b1 = b0 + (size_t)16 * SEQL;
  const bf16_t* b2 = b0 + (size_t)32 * SEQL;
  const bf16_t* b3 = b0 + (size_t)48 * SEQL;

  v8f acc[8] = {v8f{}, v8f{}, v8f{}, v8f{}, v8f{}, v8f{}, v8f{}, v8f{}};
  for (int k0 = 0; k0 < SEQL; k0 += 32) {
    if (k0 + 64 < SEQL) {
      __builtin_prefetch(arow0 + k0 + 64, 0, 3);
      __builtin_prefetch(arow1 + k0 + 64, 0, 3);
      __builtin_prefetch(b0 + k0 + 64, 0, 3);
    }
    v16bf fa0 = load_a_frag_f32(arow0, k0, half);
    v16bf fa1 = load_a_frag_f32(arow1, k0, half);
    v16bf fb0 = load_b_frag(b0, k0, half);
    v16bf fb1 = load_b_frag(b1, k0, half);
    v16bf fb2 = load_b_frag(b2, k0, half);
    v16bf fb3 = load_b_frag(b3, k0, half);
    acc[0] = wmma_acc(fa0, fb0, acc[0]);
    acc[1] = wmma_acc(fa0, fb1, acc[1]);
    acc[2] = wmma_acc(fa0, fb2, acc[2]);
    acc[3] = wmma_acc(fa0, fb3, acc[3]);
    acc[4] = wmma_acc(fa1, fb0, acc[4]);
    acc[5] = wmma_acc(fa1, fb1, acc[5]);
    acc[6] = wmma_acc(fa1, fb2, acc[6]);
    acc[7] = wmma_acc(fa1, fb3, acc[7]);
  }

  const int bb = bh >> 4, h = bh & 15;
  const int col = lane & 15;
#pragma unroll
  for (int i = 0; i < 2; ++i) {
#pragma unroll
    for (int r2 = 0; r2 < 8; ++r2) {
      const size_t base = ((size_t)bb * SEQL + m0 + i * 16 + r2 + half * 8) * DMODEL +
                          h * HDIM + col;
      ctx[base + 0]  = (bf16_t)acc[i * 4 + 0][r2];
      ctx[base + 16] = (bf16_t)acc[i * 4 + 1][r2];
      ctx[base + 32] = (bf16_t)acc[i * 4 + 2][r2];
      ctx[base + 48] = (bf16_t)acc[i * 4 + 3][r2];
    }
  }
}

// ---------------- stage 5: out = ctx Wo + bo ----------------------------
// grid (DMODEL/128, MTOK/32), block 128 (4 waves, each a 32x32 tile)
__global__ void out_proj_kernel(const bf16_t* __restrict__ ctx,
                                const bf16_t* __restrict__ woT,
                                const float* __restrict__ bo,
                                float* __restrict__ out) {
  const int lane = threadIdx.x & 31;
  const int wave = threadIdx.x >> 5;
  const int n0 = (blockIdx.x * 4 + wave) * 32;  // 0..992
  const int m0 = blockIdx.y * 32;               // 0..4064

  v8f acc[4] = {v8f{}, v8f{}, v8f{}, v8f{}};
  wmma_tile_2x2(ctx + (size_t)m0 * DMODEL, DMODEL,
                woT + (size_t)n0 * DMODEL, DMODEL, DMODEL, lane, acc);

  const int half = lane >> 4, col = lane & 15;
#pragma unroll
  for (int i = 0; i < 2; ++i) {
#pragma unroll
    for (int j = 0; j < 2; ++j) {
      const v8f a = acc[i * 2 + j];
      const int n = n0 + j * 16 + col;
      const float bz = bo[n];
#pragma unroll
      for (int r = 0; r < 8; ++r)
        out[(size_t)(m0 + i * 16 + r + half * 8) * DMODEL + n] = a[r] + bz;
    }
  }
}

// ---------------- host launcher ----------------------------------------
extern "C" void kernel_launch(void* const* d_in, const int* in_sizes, int n_in,
                              void* d_out, int out_size, void* d_ws, size_t ws_size,
                              hipStream_t stream) {
  const float* x  = (const float*)d_in[0];
  const float* wq = (const float*)d_in[1];
  const float* bq = (const float*)d_in[2];
  const float* wk = (const float*)d_in[3];
  const float* bk = (const float*)d_in[4];
  const float* wv = (const float*)d_in[5];
  const float* bv = (const float*)d_in[6];
  const float* wo = (const float*)d_in[7];
  const float* bo = (const float*)d_in[8];

  float* out  = (float*)d_out;
  float* attn = out + (size_t)MTOK * DMODEL;  // [BH, L, L] fp32

  // workspace carve-out (256B aligned)
  char* wsp = (char*)d_ws;
  auto carve = [&](size_t bytes) {
    char* p = wsp;
    wsp += (bytes + 255) & ~(size_t)255;
    return p;
  };
  bf16_t* xb  = (bf16_t*)carve((size_t)MTOK * DMODEL * sizeof(bf16_t));
  bf16_t* wqT = (bf16_t*)carve((size_t)DMODEL * DMODEL * sizeof(bf16_t));
  bf16_t* wkT = (bf16_t*)carve((size_t)DMODEL * DMODEL * sizeof(bf16_t));
  bf16_t* wvT = (bf16_t*)carve((size_t)DMODEL * DMODEL * sizeof(bf16_t));
  bf16_t* woT = (bf16_t*)carve((size_t)DMODEL * DMODEL * sizeof(bf16_t));
  bf16_t* Qb  = (bf16_t*)carve((size_t)BH * SEQL * HDIM * sizeof(bf16_t));
  bf16_t* Kb  = (bf16_t*)carve((size_t)BH * SEQL * HDIM * sizeof(bf16_t));
  bf16_t* Vt  = (bf16_t*)carve((size_t)BH * HDIM * SEQL * sizeof(bf16_t));
  bf16_t* ctx = (bf16_t*)carve((size_t)MTOK * DMODEL * sizeof(bf16_t));
  (void)ws_size; (void)in_sizes; (void)n_in; (void)out_size;

  // stage 0: conversions
  {
    const int n = MTOK * DMODEL;
    cvt_bf16_kernel<<<n / (256 * 4), 256, 0, stream>>>(x, xb, n);
    dim3 tb(32, 8), tg(DMODEL / 32, DMODEL / 32);
    cvt_transpose_kernel<<<tg, tb, 0, stream>>>(wq, wqT);
    cvt_transpose_kernel<<<tg, tb, 0, stream>>>(wk, wkT);
    cvt_transpose_kernel<<<tg, tb, 0, stream>>>(wv, wvT);
    cvt_transpose_kernel<<<tg, tb, 0, stream>>>(wo, woT);
  }
  // stage 1: QKV projections (32x32 tile per wave)
  qkv_proj_kernel<<<dim3(DMODEL / 128, MTOK / 32, 3), 128, 0, stream>>>(
      xb, wqT, wkT, wvT, bq, bk, bv, Qb, Kb, Vt);
  // stage 2: scores
  scores_kernel<<<dim3(SEQL / 128, SEQL / 32, BH), 128, 0, stream>>>(Qb, Kb, attn);
  // stage 3: softmax (normalizes attn in place -> final attn output)
  softmax_kernel<<<BH * SEQL, 256, 0, stream>>>(attn);
  // stage 4: ctx = P V  (2x4 blocked)
  ctx_gemm_kernel<<<dim3(SEQL / 128, BH), 128, 0, stream>>>(attn, Vt, ctx);
  // stage 5: output projection
  out_proj_kernel<<<dim3(DMODEL / 128, MTOK / 32), 128, 0, stream>>>(ctx, woT, bo, out);
}
